// Generator_1752346656900
// MI455X (gfx1250) — compile-verified, matrix operand order checked by reading, not run on hardware
//
#include <hip/hip_runtime.h>

typedef __attribute__((ext_vector_type(16))) _Float16 v16h_t;
typedef __attribute__((ext_vector_type(8)))  _Float16 v8h_t;
typedef __attribute__((ext_vector_type(8)))  float    v8f_t;
typedef __attribute__((ext_vector_type(4)))  float    v4f_t;

#define B_TOT   256
#define NSEQ    216
#define CDIM    192
#define HEADS   6
#define HD      32
#define ROWS    (B_TOT * NSEQ)      /* 55296 */
#define ROWT    (ROWS / 16)         /* 3456 */
#define QTILES  14
#define MPOS    1331
#define SCALE_Q 0.17677669529663687f  /* 32^-0.5 */
#define NEGBIG  (-1.0e30f)

__device__ __forceinline__ v8f_t wmma16(v16h_t a, v16h_t b, v8f_t c) {
  return __builtin_amdgcn_wmma_f32_16x16x32_f16(false, a, false, b, (short)0, c,
                                                false, false);
}

// A/B fragment from f16 row-major: lane supplies base p = row_ptr + kchunk0,
// chunk1 is 16 halves further.  K-chunk pattern per CDNA5 ISA 7.12.2.
__device__ __forceinline__ v16h_t ld_frag_h(const _Float16* p) {
  v8h_t c0 = *(const v8h_t*)p;
  v8h_t c1 = *(const v8h_t*)(p + 16);
  v16h_t r;
#pragma unroll
  for (int i = 0; i < 8; ++i) { r[i] = c0[i]; r[i + 8] = c1[i]; }
  return r;
}

// Same but from f32 memory with inline convert to f16.
__device__ __forceinline__ v16h_t ld_frag_f32(const float* p) {
  v4f_t a0 = *(const v4f_t*)p;
  v4f_t a1 = *(const v4f_t*)(p + 4);
  v4f_t b0 = *(const v4f_t*)(p + 16);
  v4f_t b1 = *(const v4f_t*)(p + 20);
  v16h_t r;
#pragma unroll
  for (int i = 0; i < 4; ++i) {
    r[i]      = (_Float16)a0[i];
    r[4 + i]  = (_Float16)a1[i];
    r[8 + i]  = (_Float16)b0[i];
    r[12 + i] = (_Float16)b1[i];
  }
  return r;
}

// ---------------------------------------------------------------------------
// Kernel 1: dynamic position bias MLP -> pos table (1331 x 6) f32
// ---------------------------------------------------------------------------
__device__ __forceinline__ void ln_relu12(const float* t, const float* g,
                                          const float* b, float* u) {
  float m = 0.f;
#pragma unroll
  for (int i = 0; i < 12; ++i) m += t[i];
  m *= (1.0f / 12.0f);
  float v = 0.f;
#pragma unroll
  for (int i = 0; i < 12; ++i) { float d = t[i] - m; v += d * d; }
  v *= (1.0f / 12.0f);
  float inv = rsqrtf(v + 1e-5f);
#pragma unroll
  for (int i = 0; i < 12; ++i) {
    float val = (t[i] - m) * inv * g[i] + b[i];
    u[i] = val > 0.f ? val : 0.f;
  }
}

__global__ __launch_bounds__(128) void dpb_kernel(
    const float* __restrict__ ppw, const float* __restrict__ ppb,
    const float* __restrict__ g1, const float* __restrict__ b1,
    const float* __restrict__ w1, const float* __restrict__ bb1,
    const float* __restrict__ g2, const float* __restrict__ b2,
    const float* __restrict__ w2, const float* __restrict__ bb2,
    const float* __restrict__ g3, const float* __restrict__ b3,
    const float* __restrict__ w3, const float* __restrict__ bb3,
    float* __restrict__ pos) {
  int m = blockIdx.x * blockDim.x + threadIdx.x;
  if (m >= MPOS) return;
  float c0 = (float)(m / 121) - 5.f;
  float c1 = (float)((m / 11) % 11) - 5.f;
  float c2 = (float)(m % 11) - 5.f;
  float t[12], u[12], t2[12];
#pragma unroll
  for (int j = 0; j < 12; ++j)
    t[j] = ppw[j * 3 + 0] * c0 + ppw[j * 3 + 1] * c1 + ppw[j * 3 + 2] * c2 + ppb[j];
  ln_relu12(t, g1, b1, u);
#pragma unroll
  for (int j = 0; j < 12; ++j) {
    float acc = bb1[j];
#pragma unroll
    for (int i = 0; i < 12; ++i) acc += u[i] * w1[j * 12 + i];
    t2[j] = acc;
  }
  ln_relu12(t2, g2, b2, u);
#pragma unroll
  for (int j = 0; j < 12; ++j) {
    float acc = bb2[j];
#pragma unroll
    for (int i = 0; i < 12; ++i) acc += u[i] * w2[j * 12 + i];
    t[j] = acc;
  }
  ln_relu12(t, g3, b3, u);
#pragma unroll
  for (int h = 0; h < HEADS; ++h) {
    float acc = bb3[h];
#pragma unroll
    for (int i = 0; i < 12; ++i) acc += u[i] * w3[h * 12 + i];
    pos[m * HEADS + h] = acc;
  }
}

// ---------------------------------------------------------------------------
// Kernel 2: QKV projection GEMM (f32 in, f16 WMMA, f16 out).
// Register-blocked: one wave computes a 16x64 output tile (4 accumulators
// sharing one A fragment) -> A traffic and f32->f16 convert VALU cut 4x.
// colBase=0,nColGroups=3 -> Q (scaled); colBase=192,nColGroups=6 -> K, V^T.
// ---------------------------------------------------------------------------
__global__ __launch_bounds__(256) void qkv_gemm(
    const float* __restrict__ in, const float* __restrict__ W,
    const float* __restrict__ bias, int colBase, int nColGroups,
    _Float16* __restrict__ Qh, _Float16* __restrict__ Kh,
    _Float16* __restrict__ Vt) {
  int lane = threadIdx.x & 31;
  int wid = (blockIdx.x * blockDim.x + threadIdx.x) >> 5;
  int rt = wid / nColGroups;
  int cg = wid % nColGroups;
  if (rt >= ROWT) return;
  int col16 = lane & 15;
  int kofs = (lane & 16) ? 8 : 0;
  int aRow = rt * 16 + col16;                  // A: lane holds row lane%16
  int cgBase = colBase + cg * 64;              // 64 output cols per wave
  v8f_t acc[4] = {};
#pragma unroll
  for (int kb = 0; kb < CDIM; kb += 32) {
    v16h_t a = ld_frag_f32(in + aRow * CDIM + kb + kofs);
#pragma unroll
    for (int j = 0; j < 4; ++j) {
      int bRow = cgBase + j * 16 + col16;      // B: lane holds W row (=out col)
      v16h_t b = ld_frag_f32(W + bRow * CDIM + kb + kofs);
      acc[j] = wmma16(a, b, acc[j]);
    }
  }
  int hi = (lane & 16) ? 8 : 0;
#pragma unroll
  for (int j = 0; j < 4; ++j) {
    int c = cgBase + j * 16 + col16;           // global output column 0..575
    float bv = bias[c];
    int which = c / CDIM;                      // 0=q 1=k 2=v
    int h = (c % CDIM) / HD;
    int d = c % HD;
#pragma unroll
    for (int v = 0; v < 8; ++v) {
      int r = rt * 16 + v + hi;
      int bIdx = r / NSEQ;
      int n = r % NSEQ;
      float val = acc[j][v] + bv;
      if (which == 0) {
        Qh[((bIdx * HEADS + h) * NSEQ + n) * HD + d] = (_Float16)(val * SCALE_Q);
      } else if (which == 1) {
        Kh[((bIdx * HEADS + h) * NSEQ + n) * HD + d] = (_Float16)val;
      } else {
        Vt[((bIdx * HEADS + h) * HD + d) * NSEQ + n] = (_Float16)val; // transposed
      }
    }
  }
}

// ---------------------------------------------------------------------------
// Kernel 3: fused flash-style attention.  One wave per (b, head, qtile).
// Per 32-key block: 2 S-WMMAs, bias(mask + dpb gather), online softmax,
// P staged through LDS (C-layout -> A-layout transpose), 2 PV-WMMAs.
// ---------------------------------------------------------------------------
__global__ __launch_bounds__(256) void attn_kernel(
    const _Float16* __restrict__ Q, const _Float16* __restrict__ K,
    const _Float16* __restrict__ Vt, const float* __restrict__ mask,
    const float* __restrict__ pos, _Float16* __restrict__ O) {
  __shared__ _Float16 pst[8][512];  // 8 waves x (16x32) f16 P tile
  int lane = threadIdx.x & 31;
  int wv = threadIdx.x >> 5;
  int wid = blockIdx.x * 8 + wv;
  int qt = wid % QTILES;
  int t = wid / QTILES;
  int h = t % HEADS;
  int b = t / HEADS;
  if (b >= B_TOT) return;
  int col = lane & 15;
  int kofs = (lane & 16) ? 8 : 0;
  int hi = (lane & 16) ? 8 : 0;

  const _Float16* Qb = Q + (size_t)(b * HEADS + h) * NSEQ * HD;
  const _Float16* Kb = K + (size_t)(b * HEADS + h) * NSEQ * HD;
  const _Float16* Vb = Vt + (size_t)(b * HEADS + h) * HD * NSEQ;
  const float* maskb = mask + (size_t)(b & 63) * (NSEQ * NSEQ);

  int qRowA = qt * 16 + col;
  if (qRowA > NSEQ - 1) qRowA = NSEQ - 1;
  v16h_t qa = ld_frag_h(Qb + qRowA * HD + kofs);

  float mrow[8], srow[8];
  v8f_t o0 = {}, o1 = {};
#pragma unroll
  for (int v = 0; v < 8; ++v) { mrow[v] = NEGBIG; srow[v] = 0.f; }

  for (int kb = 0; kb < 7; ++kb) {
    int kbase = kb * 32;
    int kr0 = kbase + col;       if (kr0 > NSEQ - 1) kr0 = NSEQ - 1;
    int kr1 = kbase + 16 + col;  if (kr1 > NSEQ - 1) kr1 = NSEQ - 1;
    v16h_t kf0 = ld_frag_h(Kb + kr0 * HD + kofs);
    v16h_t kf1 = ld_frag_h(Kb + kr1 * HD + kofs);
    v8f_t zero = {};
    v8f_t s0 = wmma16(qa, kf0, zero);
    v8f_t s1 = wmma16(qa, kf1, zero);

    int key0 = kbase + col;
    int key1 = key0 + 16;
#pragma unroll
    for (int v = 0; v < 8; ++v) {
      int qq = qt * 16 + v + hi;
      if (qq < NSEQ) {
        int hq = qq / 36, wq = (qq / 6) % 6, dq = qq % 6;
        if (key0 < NSEQ) {
          int hk = key0 / 36, wk = (key0 / 6) % 6, dk = key0 % 6;
          int idx = (hq - hk + 5) * 121 + (wq - wk + 5) * 11 + (dq - dk + 5);
          s0[v] += maskb[qq * NSEQ + key0] + pos[idx * HEADS + h];
        } else s0[v] = NEGBIG;
        if (key1 < NSEQ) {
          int hk = key1 / 36, wk = (key1 / 6) % 6, dk = key1 % 6;
          int idx = (hq - hk + 5) * 121 + (wq - wk + 5) * 11 + (dq - dk + 5);
          s1[v] += maskb[qq * NSEQ + key1] + pos[idx * HEADS + h];
        } else s1[v] = NEGBIG;
      } else { s0[v] = NEGBIG; s1[v] = NEGBIG; }
    }

    // online softmax update (row-wise reductions across 16 lanes)
    float e0[8], e1[8];
#pragma unroll
    for (int v = 0; v < 8; ++v) {
      float mx = fmaxf(s0[v], s1[v]);
#pragma unroll
      for (int off = 1; off < 16; off <<= 1)
        mx = fmaxf(mx, __shfl_xor(mx, off, 16));
      float mn = fmaxf(mrow[v], mx);
      float corr = __expf(mrow[v] - mn);
      float p0 = __expf(s0[v] - mn);
      float p1 = __expf(s1[v] - mn);
      float rs = p0 + p1;
#pragma unroll
      for (int off = 1; off < 16; off <<= 1)
        rs += __shfl_xor(rs, off, 16);
      mrow[v] = mn;
      srow[v] = srow[v] * corr + rs;
      o0[v] *= corr;
      o1[v] *= corr;
      e0[v] = p0;
      e1[v] = p1;
    }

    // stage P (C-layout) -> LDS -> reload in A-layout
    _Float16* ps = pst[wv];
#pragma unroll
    for (int v = 0; v < 8; ++v) {
      int rr = v + hi;
      ps[rr * 32 + col] = (_Float16)e0[v];
      ps[rr * 32 + 16 + col] = (_Float16)e1[v];
    }
    __asm__ volatile("s_wait_dscnt 0" ::: "memory");
    v16h_t pa = ld_frag_h(ps + col * 32 + kofs);

    // V^T B-fragments: column d contiguous over keys
    v16h_t vf0 = ld_frag_h(Vb + (col)*NSEQ + kbase + kofs);
    v16h_t vf1 = ld_frag_h(Vb + (16 + col) * NSEQ + kbase + kofs);
    o0 = wmma16(pa, vf0, o0);
    o1 = wmma16(pa, vf1, o1);
  }

#pragma unroll
  for (int v = 0; v < 8; ++v) {
    int qq = qt * 16 + v + hi;
    if (qq < NSEQ) {
      float inv = 1.0f / srow[v];
      size_t base = ((size_t)b * NSEQ + qq) * CDIM + h * HD;
      O[base + col] = (_Float16)(o0[v] * inv);
      O[base + 16 + col] = (_Float16)(o1[v] * inv);
    }
  }
}

// ---------------------------------------------------------------------------
// Kernel 4: output projection GEMM (f16 A, f32 W converted, f32 out).
// Register-blocked 16x64 per wave like qkv_gemm (3 col groups).
// ---------------------------------------------------------------------------
__global__ __launch_bounds__(256) void proj_gemm(
    const _Float16* __restrict__ O, const float* __restrict__ W,
    const float* __restrict__ bias, float* __restrict__ out) {
  int lane = threadIdx.x & 31;
  int wid = (blockIdx.x * blockDim.x + threadIdx.x) >> 5;
  int rt = wid / 3;
  int cg = wid % 3;
  if (rt >= ROWT) return;
  int col16 = lane & 15;
  int kofs = (lane & 16) ? 8 : 0;
  int aRow = rt * 16 + col16;
  int cgBase = cg * 64;
  v8f_t acc[4] = {};
#pragma unroll
  for (int kb = 0; kb < CDIM; kb += 32) {
    v16h_t a = ld_frag_h(O + (size_t)aRow * CDIM + kb + kofs);
#pragma unroll
    for (int j = 0; j < 4; ++j) {
      int bRow = cgBase + j * 16 + col16;
      v16h_t b = ld_frag_f32(W + bRow * CDIM + kb + kofs);
      acc[j] = wmma16(a, b, acc[j]);
    }
  }
  int hi = (lane & 16) ? 8 : 0;
#pragma unroll
  for (int j = 0; j < 4; ++j) {
    int c = cgBase + j * 16 + col16;
    float bv = bias[c];
#pragma unroll
    for (int v = 0; v < 8; ++v) {
      int r = rt * 16 + v + hi;
      out[(size_t)r * CDIM + c] = acc[j][v] + bv;
    }
  }
}

// ---------------------------------------------------------------------------
extern "C" void kernel_launch(void* const* d_in, const int* in_sizes, int n_in,
                              void* d_out, int out_size, void* d_ws,
                              size_t ws_size, hipStream_t stream) {
  (void)in_sizes; (void)n_in; (void)out_size; (void)ws_size;
  const float* x     = (const float*)d_in[0];
  const float* y     = (const float*)d_in[1];
  const float* mask  = (const float*)d_in[2];
  const float* Wqkv  = (const float*)d_in[3];
  const float* bqkv  = (const float*)d_in[4];
  const float* Wproj = (const float*)d_in[5];
  const float* bproj = (const float*)d_in[6];
  const float* ppw   = (const float*)d_in[7];
  const float* ppb   = (const float*)d_in[8];
  const float* g1    = (const float*)d_in[9];
  const float* b1    = (const float*)d_in[10];
  const float* w1    = (const float*)d_in[11];
  const float* bb1   = (const float*)d_in[12];
  const float* g2    = (const float*)d_in[13];
  const float* b2    = (const float*)d_in[14];
  const float* w2    = (const float*)d_in[15];
  const float* bb2   = (const float*)d_in[16];
  const float* g3    = (const float*)d_in[17];
  const float* b3    = (const float*)d_in[18];
  const float* w3    = (const float*)d_in[19];
  const float* bb3   = (const float*)d_in[20];
  // d_in[21..23] = H,W,D (hardcoded 6,6,6)

  char* ws = (char*)d_ws;
  float* pos = (float*)ws;                                   // 1331*6 f32
  _Float16* Qh = (_Float16*)(ws + 32768);                    // 55296*192 f16
  const size_t ELT = (size_t)ROWS * CDIM;                    // 10,616,832
  _Float16* Kh = Qh + ELT;
  _Float16* Vt = Kh + ELT;
  _Float16* Oh = Vt + ELT;

  dpb_kernel<<<(MPOS + 127) / 128, 128, 0, stream>>>(
      ppw, ppb, g1, b1, w1, bb1, g2, b2, w2, bb2, g3, b3, w3, bb3, pos);

  // Q from x (cols 0..191 = 3 groups of 64, scaled);
  // K,V from y (cols 192..575 = 6 groups of 64)
  qkv_gemm<<<ROWT * 3 / 8, 256, 0, stream>>>(x, Wqkv, bqkv, 0, 3, Qh, Kh, Vt);
  qkv_gemm<<<ROWT * 6 / 8, 256, 0, stream>>>(y, Wqkv, bqkv, CDIM, 6, Qh, Kh, Vt);

  // waves = 256 * 6 * 14 = 21504; 8 waves per block
  attn_kernel<<<(B_TOT * HEADS * QTILES) / 8, 256, 0, stream>>>(
      Qh, Kh, Vt, mask, pos, Oh);

  proj_gemm<<<ROWT * 3 / 8, 256, 0, stream>>>(Oh, Wproj, bproj, (float*)d_out);
}
